// RGCN_22952305229975
// MI455X (gfx1250) — matrix operand order repaired and checked by reading, not compiled, not run on hardware
//
#include <hip/hip_runtime.h>
#include <math.h>

// ---------------------------------------------------------------------------
// RGCN forward for MI455X (gfx1250): bf16 WMMA GEMMs, f32 accumulate.
// N=4096 nodes, R=8 relations, B=4 bases, H=256, C=4, NIDX=1024.
// k-loop unrolled 2x with ping-pong fragment buffers (no copy moves),
// loads issue ahead of the WMMAs that consume the other buffer; A prefetch.
// ---------------------------------------------------------------------------

typedef __attribute__((ext_vector_type(16))) __bf16 v16bf;
typedef __attribute__((ext_vector_type(8)))  float  v8f;

union FragU { v16bf v; uint4 q[2]; };

// ---- WMMA wrapper: D = A(16x32 bf16) x B(32x16 bf16) + C(f32) ----
__device__ inline v8f wmma_bf16(v16bf a, v16bf b, v8f c) {
  return __builtin_amdgcn_wmma_f32_16x16x32_bf16(
      /*neg_a=*/false, a, /*neg_b=*/false, b,
      /*c_mod=*/(short)0, c, /*reuse_a=*/false, /*reuse_b=*/false);
}

// A fragment (16x32) from f32 row-major memory, converted to bf16 in regs.
// Lane 0-15: row=lane, K chunks [0..7] & [16..23]; lane 16-31: [8..15] & [24..31].
__device__ inline v16bf load_a_frag_f32(const float* __restrict__ rowptr,
                                        int kbase, int lane) {
  int k0 = kbase + ((lane >> 4) & 1) * 8;
  float4 x0 = *(const float4*)(rowptr + k0);
  float4 x1 = *(const float4*)(rowptr + k0 + 4);
  float4 x2 = *(const float4*)(rowptr + k0 + 16);
  float4 x3 = *(const float4*)(rowptr + k0 + 20);
  v16bf f;
  f[0]=(__bf16)x0.x;  f[1]=(__bf16)x0.y;  f[2]=(__bf16)x0.z;  f[3]=(__bf16)x0.w;
  f[4]=(__bf16)x1.x;  f[5]=(__bf16)x1.y;  f[6]=(__bf16)x1.z;  f[7]=(__bf16)x1.w;
  f[8]=(__bf16)x2.x;  f[9]=(__bf16)x2.y;  f[10]=(__bf16)x2.z; f[11]=(__bf16)x2.w;
  f[12]=(__bf16)x3.x; f[13]=(__bf16)x3.y; f[14]=(__bf16)x3.z; f[15]=(__bf16)x3.w;
  return f;
}

// A fragment (16x32) from bf16 row-major memory (K contiguous).
__device__ inline v16bf load_a_frag_bf16(const __bf16* __restrict__ rowptr,
                                         int kbase, int lane) {
  int k0 = kbase + ((lane >> 4) & 1) * 8;
  FragU u;
  u.q[0] = *(const uint4*)(rowptr + k0);
  u.q[1] = *(const uint4*)(rowptr + k0 + 16);
  return u.v;
}

// B fragment (32x16) from bf16 memory where each column's K is contiguous.
// colptr already points at this lane's column. Lane 0-15: K [0..7] & [16..23]; 16-31: [8..15] & [24..31].
__device__ inline v16bf load_b_frag(const __bf16* __restrict__ colptr,
                                    int kbase, int lane) {
  int ks = kbase + ((lane >> 4) & 1) * 8;
  FragU u;
  u.q[0] = *(const uint4*)(colptr + ks);
  u.q[1] = *(const uint4*)(colptr + ks + 16);
  return u.v;
}

// ---------------------------------------------------------------------------
// Build transposed relation weights in bf16:
//   outT[r][h][m] = sum_b comp[r][b] * basis[b][m][h]
// basis: [4][Mdim][Hdim] f32.  outT: [8][Hdim][Mdim] bf16.
// LDS tile transpose so both global read (h-contig) and write (m-contig) coalesce.
// ---------------------------------------------------------------------------
__global__ __launch_bounds__(256) void rgcn_build_relT(
    const float* __restrict__ basis, const float* __restrict__ comp,
    __bf16* __restrict__ outT, int Mdim, int Hdim) {
  __shared__ float tile[4][32][33];
  __shared__ float compS[32];
  const int t  = threadIdx.x;
  const int tx = t & 31;          // fast index
  const int ty = t >> 5;          // 0..7
  const int m0 = blockIdx.x * 32;
  const int h0 = blockIdx.y * 32;
  if (t < 32) compS[t] = comp[t]; // [8][4]

  #pragma unroll
  for (int b = 0; b < 4; ++b)
    #pragma unroll
    for (int i = 0; i < 4; ++i) {
      int ml = ty * 4 + i;
      tile[b][ml][tx] = basis[((size_t)b * Mdim + (m0 + ml)) * Hdim + (h0 + tx)];
    }
  __syncthreads();

  #pragma unroll
  for (int r = 0; r < 8; ++r) {
    float c0 = compS[r * 4 + 0], c1 = compS[r * 4 + 1];
    float c2 = compS[r * 4 + 2], c3 = compS[r * 4 + 3];
    #pragma unroll
    for (int i = 0; i < 4; ++i) {
      int hl = ty * 4 + i;
      float v = c0 * tile[0][tx][hl] + c1 * tile[1][tx][hl]
              + c2 * tile[2][tx][hl] + c3 * tile[3][tx][hl];
      outT[((size_t)r * Hdim + (h0 + hl)) * Mdim + (m0 + tx)] = (__bf16)v;
    }
  }
}

// ---------------------------------------------------------------------------
// gc1: h1[n,h] = relu( sum_r sum_m A[r,n,m]*W1relT[r][h][m] + b1[h] )
// Block: 32 rows x 256 cols, 8 waves (2 M-subtiles x 4 N-subtiles of 64).
// A read once from HBM (f32), W1relT (16MB bf16) L2-resident.
// ---------------------------------------------------------------------------
__global__ __launch_bounds__(256) void rgcn_gc1_gemm(
    const float* __restrict__ A, const __bf16* __restrict__ W1relT,
    const float* __restrict__ b1, __bf16* __restrict__ h1bf) {
  const int tid  = threadIdx.x;
  const int lane = tid & 31;
  const int wid  = tid >> 5;
  const int waveM = wid & 1;
  const int waveN = wid >> 1;
  const int rowBase = blockIdx.x * 32 + waveM * 16;
  const int colBase = waveN * 64;
  const int rowA = rowBase + (lane & 15);

  v8f acc[4] = {};
  for (int r = 0; r < 8; ++r) {
    const float*  arow  = A + (size_t)(r * 4096 + rowA) * 4096;
    const __bf16* bbase = W1relT + (size_t)r * 256 * 4096;
    const __bf16* colptr[4];
    #pragma unroll
    for (int n = 0; n < 4; ++n)
      colptr[n] = bbase + (size_t)(colBase + n * 16 + (lane & 15)) * 4096;

    // prologue: fill buffer 0 (kk = 0)
    v16bf af0 = load_a_frag_f32(arow, 0, lane);
    v16bf bf0[4];
    #pragma unroll
    for (int n = 0; n < 4; ++n) bf0[n] = load_b_frag(colptr[n], 0, lane);

    for (int kk = 0; kk < 4096; kk += 64) {
      // fill buffer 1 (kk+32), then consume buffer 0 (kk)
      __builtin_prefetch(arow + ((kk + 256) & 4095), 0, 0);
      const int k1 = kk + 32;
      v16bf af1 = load_a_frag_f32(arow, k1, lane);
      v16bf bf1[4];
      #pragma unroll
      for (int n = 0; n < 4; ++n) bf1[n] = load_b_frag(colptr[n], k1, lane);
      #pragma unroll
      for (int n = 0; n < 4; ++n) acc[n] = wmma_bf16(af0, bf0[n], acc[n]);

      // fill buffer 0 (kk+64, wraps on last iter), then consume buffer 1
      __builtin_prefetch(arow + ((kk + 288) & 4095), 0, 0);
      const int k2 = (kk + 64) & 4095;
      af0 = load_a_frag_f32(arow, k2, lane);
      #pragma unroll
      for (int n = 0; n < 4; ++n) bf0[n] = load_b_frag(colptr[n], k2, lane);
      #pragma unroll
      for (int n = 0; n < 4; ++n) acc[n] = wmma_bf16(af1, bf1[n], acc[n]);
    }
  }

  const int hi = (lane >> 4) & 1;
  #pragma unroll
  for (int n = 0; n < 4; ++n) {
    int col = colBase + n * 16 + (lane & 15);
    float bias = b1[col];
    #pragma unroll
    for (int i = 0; i < 8; ++i) {
      int row = rowBase + i + hi * 8;
      float v = acc[n][i] + bias;
      v = v > 0.f ? v : 0.f;
      h1bf[(size_t)row * 256 + col] = (__bf16)v;
    }
  }
}

// ---------------------------------------------------------------------------
// H2rT[r][h][m] = sum_j W2relT[r][h][j] * h1[m][j]   (== (h1 @ W2rel_r)^T)
// Output written row-major [h][m] which IS the transposed layout gc2 needs.
// Block: 16 rows(h) x 512 cols(m), 8 waves x 64 cols. grid.z = r.
// ---------------------------------------------------------------------------
__global__ __launch_bounds__(256) void rgcn_h2rel_gemm(
    const __bf16* __restrict__ W2relT, const __bf16* __restrict__ h1bf,
    __bf16* __restrict__ H2rT) {
  const int tid  = threadIdx.x;
  const int lane = tid & 31;
  const int wid  = tid >> 5;
  const int r       = blockIdx.z;
  const int rowBase = blockIdx.y * 16;
  const int colBase = blockIdx.x * 512 + wid * 64;

  const __bf16* arow = W2relT + (size_t)(r * 256 + rowBase + (lane & 15)) * 256;
  const __bf16* colptr[4];
  #pragma unroll
  for (int n = 0; n < 4; ++n)
    colptr[n] = h1bf + (size_t)(colBase + n * 16 + (lane & 15)) * 256;

  v8f acc[4] = {};
  v16bf af0 = load_a_frag_bf16(arow, 0, lane);
  v16bf bf0[4];
  #pragma unroll
  for (int n = 0; n < 4; ++n) bf0[n] = load_b_frag(colptr[n], 0, lane);

  for (int kk = 0; kk < 256; kk += 64) {
    const int k1 = kk + 32;
    v16bf af1 = load_a_frag_bf16(arow, k1, lane);
    v16bf bf1[4];
    #pragma unroll
    for (int n = 0; n < 4; ++n) bf1[n] = load_b_frag(colptr[n], k1, lane);
    #pragma unroll
    for (int n = 0; n < 4; ++n) acc[n] = wmma_bf16(af0, bf0[n], acc[n]);

    const int k2 = (kk + 64) & 255;
    af0 = load_a_frag_bf16(arow, k2, lane);
    #pragma unroll
    for (int n = 0; n < 4; ++n) bf0[n] = load_b_frag(colptr[n], k2, lane);
    #pragma unroll
    for (int n = 0; n < 4; ++n) acc[n] = wmma_bf16(af1, bf1[n], acc[n]);
  }

  const int hi = (lane >> 4) & 1;
  __bf16* obase = H2rT + (size_t)r * 256 * 4096;
  #pragma unroll
  for (int n = 0; n < 4; ++n) {
    int col = colBase + n * 16 + (lane & 15);
    #pragma unroll
    for (int i = 0; i < 8; ++i) {
      int row = rowBase + i + hi * 8;
      obase[(size_t)row * 4096 + col] = (__bf16)acc[n][i];
    }
  }
}

// ---------------------------------------------------------------------------
// gc2 (gathered): h2g[i,h] = relu( b2[h] + sum_r sum_m A[r,idxs[i],m]*H2rT[r][h][m] )
// Only NIDX=1024 rows computed. Block: 16 gathered rows, 8 waves x 32 cols.
// ---------------------------------------------------------------------------
__global__ __launch_bounds__(256) void rgcn_gc2_gemm(
    const float* __restrict__ A, const __bf16* __restrict__ H2rT,
    const float* __restrict__ b2, const int* __restrict__ idxs,
    float* __restrict__ h2g) {
  const int tid  = threadIdx.x;
  const int lane = tid & 31;
  const int wid  = tid >> 5;
  const int rowBase = blockIdx.x * 16;
  const int colBase = wid * 32;
  const int myIdx = idxs[rowBase + (lane & 15)];

  v8f acc[2] = {};
  for (int r = 0; r < 8; ++r) {
    const float*  arow  = A + (size_t)(r * 4096 + myIdx) * 4096;
    const __bf16* bbase = H2rT + (size_t)r * 256 * 4096;
    const __bf16* colptr[2];
    #pragma unroll
    for (int n = 0; n < 2; ++n)
      colptr[n] = bbase + (size_t)(colBase + n * 16 + (lane & 15)) * 4096;

    v16bf af0 = load_a_frag_f32(arow, 0, lane);
    v16bf bf0[2];
    #pragma unroll
    for (int n = 0; n < 2; ++n) bf0[n] = load_b_frag(colptr[n], 0, lane);

    for (int kk = 0; kk < 4096; kk += 64) {
      __builtin_prefetch(arow + ((kk + 256) & 4095), 0, 0);
      const int k1 = kk + 32;
      v16bf af1 = load_a_frag_f32(arow, k1, lane);
      v16bf bf1[2];
      #pragma unroll
      for (int n = 0; n < 2; ++n) bf1[n] = load_b_frag(colptr[n], k1, lane);
      #pragma unroll
      for (int n = 0; n < 2; ++n) acc[n] = wmma_bf16(af0, bf0[n], acc[n]);

      __builtin_prefetch(arow + ((kk + 288) & 4095), 0, 0);
      const int k2 = (kk + 64) & 4095;
      af0 = load_a_frag_f32(arow, k2, lane);
      #pragma unroll
      for (int n = 0; n < 2; ++n) bf0[n] = load_b_frag(colptr[n], k2, lane);
      #pragma unroll
      for (int n = 0; n < 2; ++n) acc[n] = wmma_bf16(af1, bf1[n], acc[n]);
    }
  }

  const int hi = (lane >> 4) & 1;
  #pragma unroll
  for (int n = 0; n < 2; ++n) {
    int col = colBase + n * 16 + (lane & 15);
    float bias = b2[col];
    #pragma unroll
    for (int i = 0; i < 8; ++i) {
      int row = rowBase + i + hi * 8;
      float v = acc[n][i] + bias;
      h2g[(size_t)row * 256 + col] = v > 0.f ? v : 0.f;
    }
  }
}

// ---------------------------------------------------------------------------
// Head: f1 = relu(emb @ fc1_W + fc1_b); logits = f1 @ fc2_W + fc2_b; log_softmax.
// One block per labeled node. 134 MFLOP total -> plain VALU is fine.
// ---------------------------------------------------------------------------
__global__ __launch_bounds__(256) void rgcn_head(
    const float* __restrict__ h2g, const float* __restrict__ fc1W,
    const float* __restrict__ fc1b, const float* __restrict__ fc2W,
    const float* __restrict__ fc2b, float* __restrict__ out) {
  __shared__ float emb[256];
  __shared__ float f1[256];
  __shared__ float lg[4];
  const int i = blockIdx.x;
  const int t = threadIdx.x;

  emb[t] = h2g[(size_t)i * 256 + t];
  __syncthreads();

  float acc = fc1b[t];
  for (int j = 0; j < 256; ++j)
    acc = fmaf(emb[j], fc1W[(size_t)j * 256 + t], acc);
  f1[t] = acc > 0.f ? acc : 0.f;
  __syncthreads();

  if (t < 4) {
    float l = fc2b[t];
    for (int h = 0; h < 256; ++h) l = fmaf(f1[h], fc2W[h * 4 + t], l);
    lg[t] = l;
  }
  __syncthreads();

  if (t == 0) {
    float m = lg[0];
    #pragma unroll
    for (int c = 1; c < 4; ++c) m = fmaxf(m, lg[c]);
    float s = 0.f;
    #pragma unroll
    for (int c = 0; c < 4; ++c) s += expf(lg[c] - m);
    float ls = logf(s);
    #pragma unroll
    for (int c = 0; c < 4; ++c) out[i * 4 + c] = lg[c] - m - ls;
  }
}

// ---------------------------------------------------------------------------
extern "C" void kernel_launch(void* const* d_in, const int* in_sizes, int n_in,
                              void* d_out, int out_size, void* d_ws, size_t ws_size,
                              hipStream_t stream) {
  (void)in_sizes; (void)n_in; (void)out_size; (void)ws_size;
  const float* A        = (const float*)d_in[1];
  const int*   idxs     = (const int*)  d_in[2];
  const float* W1_basis = (const float*)d_in[3];
  const float* W1_comp  = (const float*)d_in[4];
  const float* b1       = (const float*)d_in[5];
  const float* W2_basis = (const float*)d_in[6];
  const float* W2_comp  = (const float*)d_in[7];
  const float* b2       = (const float*)d_in[8];
  const float* fc1W     = (const float*)d_in[9];
  const float* fc1b     = (const float*)d_in[10];
  const float* fc2W     = (const float*)d_in[11];
  const float* fc2b     = (const float*)d_in[12];
  float* out = (float*)d_out;

  // Workspace layout (bytes):
  char* w = (char*)d_ws;
  __bf16* W1relT = (__bf16*)(w);                 // [8][256][4096] bf16 = 16 MiB
  __bf16* h1bf   = (__bf16*)(w + (16u << 20));   // [4096][256]   bf16 =  2 MiB
  __bf16* W2relT = (__bf16*)(w + (18u << 20));   // [8][256][256] bf16 =  1 MiB
  __bf16* H2rT   = (__bf16*)(w + (19u << 20));   // [8][256][4096]bf16 = 16 MiB
  float*  h2g    = (float*) (w + (35u << 20));   // [1024][256]   f32  =  1 MiB

  // 1) W1relT[r][h][m] = sum_b comp1[r,b]*basis1[b][m][h]
  rgcn_build_relT<<<dim3(4096 / 32, 256 / 32), 256, 0, stream>>>(
      W1_basis, W1_comp, W1relT, 4096, 256);
  // 2) gc1 GEMM -> h1 (bf16, row-major [n][h])
  rgcn_gc1_gemm<<<dim3(4096 / 32), 256, 0, stream>>>(A, W1relT, b1, h1bf);
  // 3) W2relT[r][h][j] = sum_b comp2[r,b]*basis2[b][j][h]
  rgcn_build_relT<<<dim3(256 / 32, 256 / 32), 256, 0, stream>>>(
      W2_basis, W2_comp, W2relT, 256, 256);
  // 4) H2rT[r][h][m] = sum_j W2relT[r][h][j] * h1[m][j]
  rgcn_h2rel_gemm<<<dim3(4096 / 512, 256 / 16, 8), 256, 0, stream>>>(
      W2relT, h1bf, H2rT);
  // 5) gathered gc2 GEMM -> h2g (f32 [1024][256])
  rgcn_gc2_gemm<<<dim3(1024 / 16), 256, 0, stream>>>(A, H2rT, b2, idxs, h2g);
  // 6) MLP head + log_softmax -> out [1024][4]
  rgcn_head<<<dim3(1024), 256, 0, stream>>>(h2g, fc1W, fc1b, fc2W, fc2b, out);
}
